// ImprovedGIN_GAT_74586402063115
// MI455X (gfx1250) — matrix operand order repaired and checked by reading, not compiled, not run on hardware
//
#include <hip/hip_runtime.h>
#include <hip/hip_bf16.h>
#include <math.h>

#define N_NODES  100000
#define N_EDGES  1200000
#define N_GRAPHS 512
#define HID      64
#define HEADS    4
#define HEAD_DIM 16
#define IN_DIM   24
#define GDIM     10
#define NLAYERS  2

typedef __attribute__((ext_vector_type(16))) __bf16          v16bf;
typedef __attribute__((ext_vector_type(16))) unsigned short  v16u;
typedef __attribute__((ext_vector_type(8)))  unsigned short  v8u16;
typedef __attribute__((ext_vector_type(8)))  float           v8f;

// ---------- helpers ----------
__device__ inline unsigned short f2bf(float f) {
    unsigned u = __float_as_uint(f);
    u += 0x7FFFu + ((u >> 16) & 1u);   // round to nearest even
    return (unsigned short)(u >> 16);
}
// order-preserving float->uint key for atomicMax-based segment max
__device__ inline unsigned fkey(float f) {
    unsigned u = __float_as_uint(f);
    return (u & 0x80000000u) ? ~u : (u | 0x80000000u);
}
__device__ inline float funkey(unsigned k) {
    unsigned u = (k & 0x80000000u) ? (k & 0x7FFFFFFFu) : ~k;
    return __uint_as_float(u);
}
#define NEG_INF_KEY 0x007FFFFFu   // fkey(-inf)

__device__ inline float wave_sum(float v) {
#pragma unroll
    for (int m = 16; m >= 1; m >>= 1) v += __shfl_xor(v, m, 32);
    return v;
}

// ---------- generic fills / converts ----------
__global__ void k_fill_f32(float* p, float v, int n) {
    int i = blockIdx.x * blockDim.x + threadIdx.x;
    if (i < n) p[i] = v;
}
__global__ void k_fill_u32(unsigned* p, unsigned v, int n) {
    int i = blockIdx.x * blockDim.x + threadIdx.x;
    if (i < n) p[i] = v;
}
__global__ void k_tobf16(const float* __restrict__ src, unsigned short* __restrict__ dst, int n) {
    int i = blockIdx.x * blockDim.x + threadIdx.x;
    if (i < n) dst[i] = f2bf(src[i]);
}

// ---------- weight convert + transpose: f32 [K,Nout] -> bf16 [Nout,Kpad] ----------
__global__ void k_wt(const float* __restrict__ src, unsigned short* __restrict__ dst,
                     int K, int Nout, int Kpad) {
    int i = blockIdx.x * blockDim.x + threadIdx.x;
    if (i >= Nout * Kpad) return;
    int n = i / Kpad, k = i % Kpad;
    float v = (k < K) ? src[k * Nout + n] : 0.0f;
    dst[i] = f2bf(v);
}

// ---------- padded input build: hx[N,64] = bf16([x | g[batch] | 0]) ----------
__global__ void k_pad_input(const float* __restrict__ x, const float* __restrict__ gf,
                            const int* __restrict__ batch, unsigned short* __restrict__ hx) {
    int i = blockIdx.x * blockDim.x + threadIdx.x;
    if (i >= N_NODES * 64) return;
    int n = i >> 6, c = i & 63;
    float v = 0.0f;
    if (c < IN_DIM)              v = x[n * IN_DIM + c];
    else if (c < IN_DIM + GDIM)  v = gf[batch[n] * GDIM + (c - IN_DIM)];
    hx[i] = f2bf(v);
}

// ---------- WMMA GEMM: C[M, NT*16] = A[M,K](bf16 row-major) x WT^T (bf16 [Nout,K]) ----------
// 4 waves per block; each wave computes one 16-row strip across all NT n-tiles.
// K multiple of 32, M multiple of 16.
template <int NT>
__global__ __launch_bounds__(128)
void k_gemm_bf(const unsigned short* __restrict__ A, const unsigned short* __restrict__ WT,
               float* __restrict__ C, int K, int Mtiles) {
    const int lane = threadIdx.x & 31;
    const int mtile = blockIdx.x * 4 + (threadIdx.x >> 5);
    if (mtile >= Mtiles) return;                 // wave-uniform: EXEC stays all-1
    const int m0 = mtile * 16;
    const int hi = lane >> 4;                    // 0: lanes 0-15, 1: lanes 16-31
    const int Nout = NT * 16;

    v8f acc[NT];
#pragma unroll
    for (int t = 0; t < NT; t++) acc[t] = (v8f){};

    // A fragment pointers: lane<16 -> K k0..k0+15 ; lane>=16 -> K k0+16..k0+31
    const unsigned short* ap = A + (m0 + (lane & 15)) * K + (hi << 4);
    // B fragment pointers: lane<16 -> K k0..k0+15 ; lane>=16 -> K k0+16..k0+31
    const unsigned short* bp = WT + (lane & 15) * K + (hi << 4);

    for (int k0 = 0; k0 < K; k0 += 32) {
        v8u16 alo = *(const v8u16*)(ap + k0);
        v8u16 ahi = *(const v8u16*)(ap + k0 + 8);
        v16u  ar  = __builtin_shufflevector(alo, ahi,
                        0, 1, 2, 3, 4, 5, 6, 7, 8, 9, 10, 11, 12, 13, 14, 15);
        v16bf a = __builtin_bit_cast(v16bf, ar);
#pragma unroll
        for (int t = 0; t < NT; t++) {
            v16u  br = *(const v16u*)(bp + (size_t)(t * 16) * K + k0);
            v16bf b  = __builtin_bit_cast(v16bf, br);
            acc[t] = __builtin_amdgcn_wmma_f32_16x16x32_bf16(false, a, false, b,
                                                             (short)0, acc[t], false, false);
        }
    }
    const int mb = m0 + (hi << 3);
    const int nc = lane & 15;
#pragma unroll
    for (int t = 0; t < NT; t++)
#pragma unroll
        for (int i = 0; i < 8; i++)
            C[(size_t)(mb + i) * Nout + t * 16 + nc] = acc[t][i];
}

// ---------- row-wise LayerNorm epilogues (64 cols, one wave per row) ----------
__global__ void k_ln_relu(const float* __restrict__ X, const float* __restrict__ bias,
                          const float* __restrict__ g, const float* __restrict__ b,
                          float* __restrict__ out, int nrows) {
    int row = blockIdx.x * (blockDim.x >> 5) + (threadIdx.x >> 5);
    int lane = threadIdx.x & 31;
    if (row >= nrows) return;
    float v0 = X[row * 64 + lane]      + bias[lane];
    float v1 = X[row * 64 + lane + 32] + bias[lane + 32];
    float mean = wave_sum(v0 + v1) * (1.0f / 64.0f);
    float var  = wave_sum(v0 * v0 + v1 * v1) * (1.0f / 64.0f) - mean * mean;
    float inv  = rsqrtf(var + 1e-5f);
    float y0 = (v0 - mean) * inv * g[lane]      + b[lane];
    float y1 = (v1 - mean) * inv * g[lane + 32] + b[lane + 32];
    out[row * 64 + lane]      = fmaxf(y0, 0.0f);
    out[row * 64 + lane + 32] = fmaxf(y1, 0.0f);
}

__global__ void k_gin_post(const float* __restrict__ X, const float* __restrict__ bias,
                           const float* __restrict__ g, const float* __restrict__ b,
                           float* __restrict__ h, int nrows) {
    int row = blockIdx.x * (blockDim.x >> 5) + (threadIdx.x >> 5);
    int lane = threadIdx.x & 31;
    if (row >= nrows) return;
    float v0 = X[row * 64 + lane]      + bias[lane];
    float v1 = X[row * 64 + lane + 32] + bias[lane + 32];
    float mean = wave_sum(v0 + v1) * (1.0f / 64.0f);
    float var  = wave_sum(v0 * v0 + v1 * v1) * (1.0f / 64.0f) - mean * mean;
    float inv  = rsqrtf(var + 1e-5f);
    float y0 = fmaxf((v0 - mean) * inv * g[lane]      + b[lane], 0.0f);
    float y1 = fmaxf((v1 - mean) * inv * g[lane + 32] + b[lane + 32], 0.0f);
    h[row * 64 + lane]      += y0;     // residual
    h[row * 64 + lane + 32] += y1;
}

__global__ void k_gat_post(const float* __restrict__ X, const float* __restrict__ bias,
                           const float* __restrict__ g, const float* __restrict__ b,
                           float* __restrict__ h, int nrows) {
    int row = blockIdx.x * (blockDim.x >> 5) + (threadIdx.x >> 5);
    int lane = threadIdx.x & 31;
    if (row >= nrows) return;
    float v0 = X[row * 64 + lane]      + bias[lane];
    float v1 = X[row * 64 + lane + 32] + bias[lane + 32];
    float mean = wave_sum(v0 + v1) * (1.0f / 64.0f);
    float var  = wave_sum(v0 * v0 + v1 * v1) * (1.0f / 64.0f) - mean * mean;
    float inv  = rsqrtf(var + 1e-5f);
    float y0 = (v0 - mean) * inv * g[lane]      + b[lane];
    float y1 = (v1 - mean) * inv * g[lane + 32] + b[lane + 32];
    y0 = (y0 > 0.0f) ? y0 : (__expf(y0) - 1.0f);   // ELU
    y1 = (y1 > 0.0f) ? y1 : (__expf(y1) - 1.0f);
    h[row * 64 + lane]      += y0;
    h[row * 64 + lane + 32] += y1;
}

// ---------- GIN ----------
__global__ void k_edge_scatter(const int* __restrict__ ei, const float* __restrict__ h,
                               float* __restrict__ agg) {
    long long t = (long long)blockIdx.x * blockDim.x + threadIdx.x;
    if (t >= (long long)N_EDGES * 16) return;
    int e = (int)(t >> 4);
    int q = (int)(t & 15) << 2;
    int s = ei[e];
    int d = ei[N_EDGES + e];
    float4 v = *(const float4*)(h + s * 64 + q);
    atomicAdd(&agg[d * 64 + q + 0], v.x);
    atomicAdd(&agg[d * 64 + q + 1], v.y);
    atomicAdd(&agg[d * 64 + q + 2], v.z);
    atomicAdd(&agg[d * 64 + q + 3], v.w);
}

// out is bf16 (feeds WMMA GEMM)
__global__ void k_gin_combine(const float* __restrict__ h, const float* __restrict__ agg,
                              const float* __restrict__ epsv, int layer,
                              unsigned short* __restrict__ out) {
    int i = blockIdx.x * blockDim.x + threadIdx.x;
    if (i >= N_NODES * 64) return;
    float e = 1.0f + epsv[layer];
    out[i] = f2bf(e * h[i] + agg[i]);
}

// reads f32 GEMM output, writes bf16 activations for the next GEMM
__global__ void k_bn_relu(const float* __restrict__ z, unsigned short* __restrict__ zb,
                          const float* __restrict__ b1, const float* __restrict__ g,
                          const float* __restrict__ b, const float* __restrict__ m,
                          const float* __restrict__ v, int layer) {
    int i = blockIdx.x * blockDim.x + threadIdx.x;
    if (i >= N_NODES * 128) return;
    int c = (i & 127) + layer * 128;
    float val = z[i] + b1[c];
    val = (val - m[c]) * rsqrtf(v[c] + 1e-5f) * g[c] + b[c];
    zb[i] = f2bf(fmaxf(val, 0.0f));
}

// ---------- GAT ----------
__global__ void k_attn_scores(const float* __restrict__ xp, const float* __restrict__ asrc,
                              const float* __restrict__ adst, int layer,
                              float* __restrict__ asr, float* __restrict__ adt) {
    int i = blockIdx.x * blockDim.x + threadIdx.x;
    if (i >= N_NODES * HEADS) return;
    int n = i >> 2, hh = i & 3;
    float s1 = 0.0f, s2 = 0.0f;
#pragma unroll
    for (int d = 0; d < HEAD_DIM; d++) {
        float v = xp[n * 64 + hh * 16 + d];
        s1 += v * asrc[layer * 64 + hh * 16 + d];
        s2 += v * adst[layer * 64 + hh * 16 + d];
    }
    asr[i] = s1;
    adt[i] = s2;
}

__device__ inline void edge_sd(const int* ei, int e, int& s, int& d) {
    if (e < N_EDGES) { s = ei[e]; d = ei[N_EDGES + e]; }
    else             { s = d = e - N_EDGES; }          // self loops
}

__global__ void k_edge_max(const int* __restrict__ ei, const float* __restrict__ asr,
                           const float* __restrict__ adt, unsigned* __restrict__ nmax) {
    int t = blockIdx.x * blockDim.x + threadIdx.x;
    if (t >= (N_EDGES + N_NODES) * HEADS) return;
    int e = t >> 2, hh = t & 3, s, d;
    edge_sd(ei, e, s, d);
    float x = asr[s * 4 + hh] + adt[d * 4 + hh];
    x = (x > 0.0f) ? x : 0.2f * x;                     // leaky relu
    atomicMax(&nmax[d * 4 + hh], fkey(x));
}

__global__ void k_edge_sum(const int* __restrict__ ei, const float* __restrict__ asr,
                           const float* __restrict__ adt, const unsigned* __restrict__ nmax,
                           float* __restrict__ nsum) {
    int t = blockIdx.x * blockDim.x + threadIdx.x;
    if (t >= (N_EDGES + N_NODES) * HEADS) return;
    int e = t >> 2, hh = t & 3, s, d;
    edge_sd(ei, e, s, d);
    float x = asr[s * 4 + hh] + adt[d * 4 + hh];
    x = (x > 0.0f) ? x : 0.2f * x;
    float mv = funkey(nmax[d * 4 + hh]);
    atomicAdd(&nsum[d * 4 + hh], __expf(x - mv));
}

__global__ void k_edge_agg(const int* __restrict__ ei, const float* __restrict__ asr,
                           const float* __restrict__ adt, const unsigned* __restrict__ nmax,
                           const float* __restrict__ nsum, const float* __restrict__ xp,
                           float* __restrict__ out) {
    int t = blockIdx.x * blockDim.x + threadIdx.x;
    if (t >= (N_EDGES + N_NODES) * HEADS) return;
    int e = t >> 2, hh = t & 3, s, dn;
    edge_sd(ei, e, s, dn);
    float x = asr[s * 4 + hh] + adt[dn * 4 + hh];
    x = (x > 0.0f) ? x : 0.2f * x;
    float mv = funkey(nmax[dn * 4 + hh]);
    float alpha = __expf(x - mv) / nsum[dn * 4 + hh];
#pragma unroll
    for (int d = 0; d < HEAD_DIM; d++)
        atomicAdd(&out[dn * 64 + hh * 16 + d], alpha * xp[s * 64 + hh * 16 + d]);
}

// ---------- pooling ----------
__global__ void k_pool(const float* __restrict__ gin_out, const float* __restrict__ h,
                       const int* __restrict__ batch, float* __restrict__ addp,
                       float* __restrict__ sgat, unsigned* __restrict__ maxb,
                       float* __restrict__ cnt) {
    int i = blockIdx.x * blockDim.x + threadIdx.x;
    if (i >= N_NODES * 64) return;
    int n = i >> 6, c = i & 63;
    int g = batch[n];
    atomicAdd(&addp[g * 64 + c], gin_out[i]);
    atomicAdd(&sgat[g * 64 + c], h[i]);
    atomicMax(&maxb[g * 64 + c], fkey(h[i]));
    if (c == 0) atomicAdd(&cnt[g], 1.0f);
}

// writes bf16 pooled features directly (feeds fusion GEMM)
__global__ void k_pool_fin(const float* __restrict__ addp, const float* __restrict__ sgat,
                           const unsigned* __restrict__ maxb, const float* __restrict__ cnt,
                           unsigned short* __restrict__ pooled) {
    int i = blockIdx.x * blockDim.x + threadIdx.x;
    if (i >= N_GRAPHS * 192) return;
    int g = i / 192, c = i % 192;
    float v;
    if (c < 64)       v = addp[g * 64 + c];
    else if (c < 128) v = sgat[g * 64 + (c - 64)] / fmaxf(cnt[g], 1.0f);
    else {
        float mx = funkey(maxb[g * 64 + (c - 128)]);
        v = (cnt[g] > 0.0f) ? mx : 0.0f;
    }
    pooled[i] = f2bf(v);
}

__global__ void k_bias_relu(float* __restrict__ z, const float* __restrict__ b,
                            int ncols, int n) {
    int i = blockIdx.x * blockDim.x + threadIdx.x;
    if (i < n) z[i] = fmaxf(z[i] + b[i % ncols], 0.0f);
}

__global__ void k_pred(const float* __restrict__ r2, const float* __restrict__ w1,
                       const float* __restrict__ b1, const float* __restrict__ w2,
                       const float* __restrict__ b2, float* __restrict__ out) {
    int g = blockIdx.x * blockDim.x + threadIdx.x;
    if (g >= N_GRAPHS) return;
    float acc = b2[0];
#pragma unroll
    for (int j = 0; j < 16; j++) {
        float s = b1[j];
        for (int k = 0; k < 32; k++) s += r2[g * 32 + k] * w1[k * 16 + j];
        acc += fmaxf(s, 0.0f) * w2[j];
    }
    out[g] = acc;
}

// ====================== launch ======================
static inline int cdiv(long long a, int b) { return (int)((a + b - 1) / b); }

extern "C" void kernel_launch(void* const* d_in, const int* in_sizes, int n_in,
                              void* d_out, int out_size, void* d_ws, size_t ws_size,
                              hipStream_t stream) {
    (void)in_sizes; (void)n_in; (void)out_size; (void)ws_size;
    const float* x       = (const float*)d_in[0];
    const float* gfeat   = (const float*)d_in[1];
    const int*   ei      = (const int*)d_in[2];
    const int*   batch   = (const int*)d_in[3];
    const float* in_w    = (const float*)d_in[4];
    const float* in_b    = (const float*)d_in[5];
    const float* in_ln_g = (const float*)d_in[6];
    const float* in_ln_b = (const float*)d_in[7];
    const float* gin_eps = (const float*)d_in[8];
    const float* gin_w1  = (const float*)d_in[9];
    const float* gin_b1  = (const float*)d_in[10];
    const float* gin_bn_g= (const float*)d_in[11];
    const float* gin_bn_b= (const float*)d_in[12];
    const float* gin_bn_m= (const float*)d_in[13];
    const float* gin_bn_v= (const float*)d_in[14];
    const float* gin_w2  = (const float*)d_in[15];
    const float* gin_b2  = (const float*)d_in[16];
    const float* gin_ln_g= (const float*)d_in[17];
    const float* gin_ln_b= (const float*)d_in[18];
    const float* gat_w   = (const float*)d_in[19];
    const float* gat_asrc= (const float*)d_in[20];
    const float* gat_adst= (const float*)d_in[21];
    const float* gat_bias= (const float*)d_in[22];
    const float* gat_ln_g= (const float*)d_in[23];
    const float* gat_ln_b= (const float*)d_in[24];
    const float* fus_w1  = (const float*)d_in[25];
    const float* fus_b1  = (const float*)d_in[26];
    const float* fus_ln_g= (const float*)d_in[27];
    const float* fus_ln_b= (const float*)d_in[28];
    const float* fus_w2  = (const float*)d_in[29];
    const float* fus_b2  = (const float*)d_in[30];
    const float* pred_w1 = (const float*)d_in[31];
    const float* pred_b1 = (const float*)d_in[32];
    const float* pred_w2 = (const float*)d_in[33];
    const float* pred_b2 = (const float*)d_in[34];
    float* out = (float*)d_out;

    // ---- workspace carve-up ----
    size_t off = 0;
    auto alloc = [&](size_t bytes) -> void* {
        void* p = (char*)d_ws + off;
        off += (bytes + 255) & ~(size_t)255;
        return p;
    };
    float* h      = (float*)alloc((size_t)N_NODES * 64 * 4);
    float* tmp64  = (float*)alloc((size_t)N_NODES * 64 * 4);
    float* agg    = (float*)alloc((size_t)N_NODES * 64 * 4);   // aggregation / GAT out
    float* z128   = (float*)alloc((size_t)N_NODES * 128 * 4);
    float* ginout = (float*)alloc((size_t)N_NODES * 64 * 4);
    float* xp     = (float*)alloc((size_t)N_NODES * 64 * 4);
    float* asr    = (float*)alloc((size_t)N_NODES * 4 * 4);
    float* adt    = (float*)alloc((size_t)N_NODES * 4 * 4);
    unsigned* nmax= (unsigned*)alloc((size_t)N_NODES * 4 * 4);
    float* nsum   = (float*)alloc((size_t)N_NODES * 4 * 4);
    float* addp   = (float*)alloc((size_t)N_GRAPHS * 64 * 4);
    float* sgat   = (float*)alloc((size_t)N_GRAPHS * 64 * 4);
    unsigned* maxb= (unsigned*)alloc((size_t)N_GRAPHS * 64 * 4);
    float* cnt    = (float*)alloc((size_t)N_GRAPHS * 4);
    float* r1     = (float*)alloc((size_t)N_GRAPHS * 64 * 4);
    float* r2     = (float*)alloc((size_t)N_GRAPHS * 32 * 4);
    // bf16 activation buffers (WMMA GEMM inputs)
    unsigned short* actb   = (unsigned short*)alloc((size_t)N_NODES * 64 * 2);  // hx / GIN combine
    unsigned short* z128b  = (unsigned short*)alloc((size_t)N_NODES * 128 * 2);
    unsigned short* hb     = (unsigned short*)alloc((size_t)N_NODES * 64 * 2);
    unsigned short* pooledb= (unsigned short*)alloc((size_t)N_GRAPHS * 192 * 2);
    unsigned short* r1b    = (unsigned short*)alloc((size_t)N_GRAPHS * 64 * 2);
    // bf16 weights (transposed, K-padded)
    unsigned short* wInT  = (unsigned short*)alloc(64 * 64 * 2);
    unsigned short* wG1T  = (unsigned short*)alloc(NLAYERS * 128 * 64 * 2);
    unsigned short* wG2T  = (unsigned short*)alloc(NLAYERS * 64 * 128 * 2);
    unsigned short* wGatT = (unsigned short*)alloc(NLAYERS * 64 * 64 * 2);
    unsigned short* wF1T  = (unsigned short*)alloc(64 * 192 * 2);
    unsigned short* wF2T  = (unsigned short*)alloc(32 * 64 * 2);

    const int BT = 256;
    const int MT_N = N_NODES / 16;   // 6250 node m-tiles
    const int MT_G = N_GRAPHS / 16;  // 32 graph m-tiles

    // ---- weight conversion ----
    k_wt<<<cdiv(64 * 64, BT), BT, 0, stream>>>(in_w, wInT, IN_DIM + GDIM, 64, 64);
    for (int l = 0; l < NLAYERS; l++) {
        k_wt<<<cdiv(128 * 64, BT), BT, 0, stream>>>(gin_w1 + (size_t)l * 64 * 128,
                                                    wG1T + (size_t)l * 128 * 64, 64, 128, 64);
        k_wt<<<cdiv(64 * 128, BT), BT, 0, stream>>>(gin_w2 + (size_t)l * 128 * 64,
                                                    wG2T + (size_t)l * 64 * 128, 128, 64, 128);
        k_wt<<<cdiv(64 * 64, BT), BT, 0, stream>>>(gat_w + (size_t)l * 64 * 64,
                                                   wGatT + (size_t)l * 64 * 64, 64, 64, 64);
    }
    k_wt<<<cdiv(64 * 192, BT), BT, 0, stream>>>(fus_w1, wF1T, 192, 64, 192);
    k_wt<<<cdiv(32 * 64, BT), BT, 0, stream>>>(fus_w2, wF2T, 64, 32, 64);

    // ---- input projection ----
    k_pad_input<<<cdiv((long long)N_NODES * 64, BT), BT, 0, stream>>>(x, gfeat, batch, actb);
    k_gemm_bf<4><<<cdiv(MT_N, 4), 128, 0, stream>>>(actb, wInT, tmp64, 64, MT_N);
    k_ln_relu<<<cdiv(N_NODES, 8), 256, 0, stream>>>(tmp64, in_b, in_ln_g, in_ln_b, h, N_NODES);

    // ---- GIN stack ----
    for (int l = 0; l < NLAYERS; l++) {
        k_fill_f32<<<cdiv((long long)N_NODES * 64, BT), BT, 0, stream>>>(agg, 0.0f, N_NODES * 64);
        k_edge_scatter<<<cdiv((long long)N_EDGES * 16, BT), BT, 0, stream>>>(ei, h, agg);
        k_gin_combine<<<cdiv((long long)N_NODES * 64, BT), BT, 0, stream>>>(h, agg, gin_eps, l, actb);
        k_gemm_bf<8><<<cdiv(MT_N, 4), 128, 0, stream>>>(actb, wG1T + (size_t)l * 128 * 64, z128, 64, MT_N);
        k_bn_relu<<<cdiv((long long)N_NODES * 128, BT), BT, 0, stream>>>(z128, z128b, gin_b1, gin_bn_g,
                                                                         gin_bn_b, gin_bn_m, gin_bn_v, l);
        k_gemm_bf<4><<<cdiv(MT_N, 4), 128, 0, stream>>>(z128b, wG2T + (size_t)l * 64 * 128, tmp64, 128, MT_N);
        k_gin_post<<<cdiv(N_NODES, 8), 256, 0, stream>>>(tmp64, gin_b2 + l * 64,
                                                         gin_ln_g + l * 64, gin_ln_b + l * 64, h, N_NODES);
    }
    hipMemcpyAsync(ginout, h, (size_t)N_NODES * 64 * 4, hipMemcpyDeviceToDevice, stream);

    // ---- GAT stack ----
    const int EH = (N_EDGES + N_NODES) * HEADS;
    for (int l = 0; l < NLAYERS; l++) {
        k_tobf16<<<cdiv((long long)N_NODES * 64, BT), BT, 0, stream>>>(h, hb, N_NODES * 64);
        k_gemm_bf<4><<<cdiv(MT_N, 4), 128, 0, stream>>>(hb, wGatT + (size_t)l * 64 * 64, xp, 64, MT_N);
        k_attn_scores<<<cdiv((long long)N_NODES * HEADS, BT), BT, 0, stream>>>(xp, gat_asrc, gat_adst, l, asr, adt);
        k_fill_u32<<<cdiv((long long)N_NODES * 4, BT), BT, 0, stream>>>(nmax, NEG_INF_KEY, N_NODES * 4);
        k_fill_f32<<<cdiv((long long)N_NODES * 4, BT), BT, 0, stream>>>(nsum, 0.0f, N_NODES * 4);
        k_edge_max<<<cdiv(EH, BT), BT, 0, stream>>>(ei, asr, adt, nmax);
        k_edge_sum<<<cdiv(EH, BT), BT, 0, stream>>>(ei, asr, adt, nmax, nsum);
        k_fill_f32<<<cdiv((long long)N_NODES * 64, BT), BT, 0, stream>>>(agg, 0.0f, N_NODES * 64);
        k_edge_agg<<<cdiv(EH, BT), BT, 0, stream>>>(ei, asr, adt, nmax, nsum, xp, agg);
        k_gat_post<<<cdiv(N_NODES, 8), 256, 0, stream>>>(agg, gat_bias + l * 64,
                                                         gat_ln_g + l * 64, gat_ln_b + l * 64, h, N_NODES);
    }

    // ---- pooling ----
    k_fill_f32<<<cdiv(N_GRAPHS * 64, BT), BT, 0, stream>>>(addp, 0.0f, N_GRAPHS * 64);
    k_fill_f32<<<cdiv(N_GRAPHS * 64, BT), BT, 0, stream>>>(sgat, 0.0f, N_GRAPHS * 64);
    k_fill_u32<<<cdiv(N_GRAPHS * 64, BT), BT, 0, stream>>>(maxb, NEG_INF_KEY, N_GRAPHS * 64);
    k_fill_f32<<<cdiv(N_GRAPHS, BT), BT, 0, stream>>>(cnt, 0.0f, N_GRAPHS);
    k_pool<<<cdiv((long long)N_NODES * 64, BT), BT, 0, stream>>>(ginout, h, batch, addp, sgat, maxb, cnt);
    k_pool_fin<<<cdiv(N_GRAPHS * 192, BT), BT, 0, stream>>>(addp, sgat, maxb, cnt, pooledb);

    // ---- fusion + prediction heads ----
    k_gemm_bf<4><<<cdiv(MT_G, 4), 128, 0, stream>>>(pooledb, wF1T, r1, 192, MT_G);
    k_ln_relu<<<cdiv(N_GRAPHS, 8), 256, 0, stream>>>(r1, fus_b1, fus_ln_g, fus_ln_b, r1, N_GRAPHS);
    k_tobf16<<<cdiv(N_GRAPHS * 64, BT), BT, 0, stream>>>(r1, r1b, N_GRAPHS * 64);
    k_gemm_bf<2><<<cdiv(MT_G, 4), 128, 0, stream>>>(r1b, wF2T, r2, 64, MT_G);
    k_bias_relu<<<cdiv(N_GRAPHS * 32, BT), BT, 0, stream>>>(r2, fus_b2, 32, N_GRAPHS * 32);
    k_pred<<<cdiv(N_GRAPHS, BT), BT, 0, stream>>>(r2, pred_w1, pred_b1, pred_w2, pred_b2, out);
}
